// Generator_39101382263302
// MI455X (gfx1250) — matrix operand order repaired and checked by reading, not compiled, not run on hardware
//
#include <hip/hip_runtime.h>
#include <hip/hip_bf16.h>
#include <math.h>

// ---------------- problem constants ----------------
#define VOCAB 32000
#define BATCH 64
#define EMB   128
#define HID   256
#define SEQ   64
#define ROWS  (SEQ * BATCH)   // 4096 (t-major rows: row = t*BATCH + b)
#define GCOLS (4 * HID)       // 1024 (i|f|o|c gate columns)
#define NT_G  (GCOLS / 16)    // 64
#define KT_X  (EMB / 32)      // 4
#define KT_H  (HID / 32)      // 8
#define NT_V  (VOCAB / 16)    // 2000
#define CHUNKS 25             // stats pass: 2000 ntiles = 25 chunks * 8 waves * 10 tiles
#define NT_PER_WAVE 10

typedef __attribute__((ext_vector_type(16))) _Float16 v16h;
typedef __attribute__((ext_vector_type(8)))  _Float16 v8h;
typedef __attribute__((ext_vector_type(8)))  float    v8f;

// D = A*B + C, f16 inputs, f32 accumulate (CDNA5 V_WMMA_F32_16X16X32_F16)
__device__ __forceinline__ v8f wmma16(v16h a, v16h b, v8f c) {
  return __builtin_amdgcn_wmma_f32_16x16x32_f16(false, a, false, b, (short)0, c,
                                                false, false);
}

__device__ __forceinline__ v16h join16(v8h lo, v8h hi) {
  v16h r;
#pragma unroll
  for (int i = 0; i < 8; ++i) { r[i] = lo[i]; r[i + 8] = hi[i]; }
  return r;
}

// A-fragment (16x32 f16) from a row-major [M x ld] f16 matrix.
// Lane L: row = mbase + (L&15); K = k32 + (L>>4)*16 + {0..15}
// -> two contiguous 16B loads per lane.
__device__ __forceinline__ v16h load_a(const _Float16* p, int ld, int mbase,
                                       int k32, int lane) {
  const _Float16* q = p + (size_t)(mbase + (lane & 15)) * ld + k32 + ((lane >> 4) << 4);
  return join16(*(const v8h*)q, *(const v8h*)(q + 8));
}

// online-softmax combine: (m,s) <- (m,s) merged with (m2,s2)
__device__ __forceinline__ void sm_merge(float& m, float& s, float m2, float s2) {
  float nm = fmaxf(m, m2);
  s = s * __expf(m - nm) + s2 * __expf(m2 - nm);
  m = nm;
}

// =====================================================================
// K0: build xs[t][b][:] = emb[t==0 ? START : input_x[b][t-1]] as f16
// =====================================================================
__global__ void k_embed(const int* __restrict__ input_x,
                        const float* __restrict__ emb,
                        _Float16* __restrict__ XS) {
  int idx = blockIdx.x * blockDim.x + threadIdx.x;
  if (idx >= ROWS * EMB) return;
  int e   = idx & (EMB - 1);
  int row = idx >> 7;              // EMB = 128
  int t = row / BATCH, b = row % BATCH;
  int tok = (t == 0) ? 0 : input_x[b * SEQ + (t - 1)];   // START = 0
  XS[idx] = (_Float16)emb[(size_t)tok * EMB + e];
}

// =====================================================================
// Pack 4 gate weight matrices [K x HID] f32 into WMMA B-fragment order:
// dst[(nt*KT + kt)*32 + lane][i] = W_g[kt*32 + (lane>>4)*16 + i][n & 255]
// (gate g = n>>8, n = nt*16 + (lane&15)). One-time; GEMMs then do
// contiguous 32B/lane loads.
// =====================================================================
__global__ void k_pack_gates(const float* __restrict__ W0, const float* __restrict__ W1,
                             const float* __restrict__ W2, const float* __restrict__ W3,
                             int K, _Float16* __restrict__ dst) {
  int KT = K >> 5;
  int idx = blockIdx.x * blockDim.x + threadIdx.x;      // NT_G * KT * 32
  if (idx >= NT_G * KT * 32) return;
  int lane = idx & 31;
  int kt = (idx >> 5) % KT;
  int nt = (idx >> 5) / KT;
  int n  = nt * 16 + (lane & 15);
  const float* W = (n < 256) ? W0 : (n < 512) ? W1 : (n < 768) ? W2 : W3;
  int nc = n & 255;
  _Float16* o = dst + ((size_t)(nt * KT + kt) * 32 + lane) * 16;
  int kbase = kt * 32 + ((lane >> 4) << 4);
#pragma unroll
  for (int i = 0; i < 16; ++i)
    o[i] = (_Float16)W[(size_t)(kbase + i) * HID + nc];
}

// Same packing for Wo [HID x VOCAB] f32 -> 16 MB f16 (L2-resident).
__global__ void k_pack_wo(const float* __restrict__ Wo, _Float16* __restrict__ dst) {
  int idx = blockIdx.x * blockDim.x + threadIdx.x;      // NT_V * KT_H * 32
  if (idx >= NT_V * KT_H * 32) return;
  int lane = idx & 31;
  int kt = (idx >> 5) % KT_H;
  int nt = (idx >> 5) / KT_H;
  int n  = nt * 16 + (lane & 15);
  _Float16* o = dst + ((size_t)(nt * KT_H + kt) * 32 + lane) * 16;
  int kbase = kt * 32 + ((lane >> 4) << 4);
#pragma unroll
  for (int i = 0; i < 16; ++i)
    o[i] = (_Float16)Wo[(size_t)(kbase + i) * VOCAB + n];
}

// =====================================================================
// K1: gates_x[t*B+b][0:1024] = xs @ [Wi|Wf|Wog|Wc] + [bi|bf|bog|bc]
// One 16x16 tile per wave, K = 128 -> 4 WMMAs.
// =====================================================================
__global__ void k_gates_x(const _Float16* __restrict__ XS,
                          const _Float16* __restrict__ Wx_pk,
                          const float* __restrict__ bi, const float* __restrict__ bf,
                          const float* __restrict__ bog, const float* __restrict__ bc,
                          float* __restrict__ gx) {
  int wave = (blockIdx.x * blockDim.x + threadIdx.x) >> 5;  // 0..16383
  int lane = threadIdx.x & 31;
  int nt = wave & (NT_G - 1);
  int mt = wave >> 6;
  v8f acc = {};
#pragma unroll
  for (int kt = 0; kt < KT_X; ++kt) {
    v16h a = load_a(XS, EMB, mt * 16, kt * 32, lane);
    v16h b = *(const v16h*)(Wx_pk + ((size_t)(nt * KT_X + kt) * 32 + lane) * 16);
    acc = wmma16(a, b, acc);
  }
  int n = nt * 16 + (lane & 15);
  const float* bias = (n < 256) ? bi : (n < 512) ? bf : (n < 768) ? bog : bc;
  float bv = bias[n & 255];
  int m0 = mt * 16 + ((lane >> 4) << 3);
#pragma unroll
  for (int r = 0; r < 8; ++r)
    gx[(size_t)(m0 + r) * GCOLS + n] = acc[r] + bv;
}

// =====================================================================
// K2: sequential LSTM recurrence. One workgroup (32 wave32 = 8/SIMD on
// one WGP). h (f16, double-buffered) and c (f32) live in LDS (128 KB of
// 320 KB). Per step each wave owns two (batch-tile, hcol-tile) tasks and
// computes all 4 gates for the tile -> gate fusion is lane-local.
// =====================================================================
__global__ void __launch_bounds__(1024)
k_recurrence(const float* __restrict__ gx, const _Float16* __restrict__ U_pk,
             _Float16* __restrict__ H_all) {
  __shared__ __align__(16) _Float16 hbuf[2][BATCH][HID];  // 2 x 32 KB
  __shared__ __align__(16) float    cbuf[BATCH][HID];     // 64 KB
  int tid = threadIdx.x;
  for (int i = tid; i < BATCH * HID; i += 1024) {
    (&hbuf[0][0][0])[i] = (_Float16)0.f;
    (&cbuf[0][0])[i] = 0.f;
  }
  __syncthreads();

  int wave = tid >> 5, lane = tid & 31;
  for (int t = 0; t < SEQ; ++t) {
    const int cur = t & 1, nxt = cur ^ 1;
    for (int task = wave; task < 64; task += 32) {
      int mt = task >> 4;      // batch tile 0..3
      int nt = task & 15;      // hid-col tile 0..15
      v8f acc[4] = {{}, {}, {}, {}};
#pragma unroll
      for (int kt = 0; kt < KT_H; ++kt) {
        // A fragment of h_t from LDS: two ds_load_b128 per lane
        const _Float16* q =
            &hbuf[cur][mt * 16 + (lane & 15)][kt * 32 + ((lane >> 4) << 4)];
        v16h a = join16(*(const v8h*)q, *(const v8h*)(q + 8));
#pragma unroll
        for (int g = 0; g < 4; ++g) {
          v16h b = *(const v16h*)(
              U_pk + ((size_t)((g * 16 + nt) * KT_H + kt) * 32 + lane) * 16);
          acc[g] = wmma16(a, b, acc[g]);
        }
      }
      int ncol = nt * 16 + (lane & 15);
      int m0 = mt * 16 + ((lane >> 4) << 3);
#pragma unroll
      for (int r = 0; r < 8; ++r) {
        int b = m0 + r;
        const float* g0 = gx + (size_t)(t * BATCH + b) * GCOLS;
        float pi = acc[0][r] + g0[0 * HID + ncol];
        float pf = acc[1][r] + g0[1 * HID + ncol];
        float po = acc[2][r] + g0[2 * HID + ncol];
        float pc = acc[3][r] + g0[3 * HID + ncol];
        float iv = 1.f / (1.f + __expf(-pi));
        float fv = 1.f / (1.f + __expf(-pf));
        float ov = 1.f / (1.f + __expf(-po));
        float cn = fv * cbuf[b][ncol] + iv * tanhf(pc);
        cbuf[b][ncol] = cn;
        float hn = ov * tanhf(cn);
        hbuf[nxt][b][ncol] = (_Float16)hn;
        H_all[(size_t)(t * BATCH + b) * HID + ncol] = (_Float16)hn;
      }
    }
    __syncthreads();
  }
}

// =====================================================================
// K3a: softmax STATS pass, logits never written. Each wave owns a 16-row
// tile and 10 column tiles; WMMA computes the tile, lanes keep per-row
// online (max, sum-exp) in registers, butterfly shfl_xor combine at the
// end, wave partials merged via 1KB LDS -> 25 chunk partials per row.
// Wo_pk stays in L2; d_out untouched (saves ~1.5 GB of HBM traffic).
// =====================================================================
__global__ void __launch_bounds__(256)
k_logit_stats(const _Float16* __restrict__ H_all,
              const _Float16* __restrict__ Wo_pk,
              const float* __restrict__ bo,
              float* __restrict__ pm, float* __restrict__ ps) {
  __shared__ float lm[8][16], lsum[8][16];
  int lane = threadIdx.x & 31;
  int wv   = threadIdx.x >> 5;          // 0..7
  int mt    = blockIdx.y;               // 0..255
  int chunk = blockIdx.x;               // 0..24
  int nt0 = (chunk * 8 + wv) * NT_PER_WAVE;

  // A fragments for this row tile, resident in registers for all 10 ntiles
  v16h afrag[KT_H];
#pragma unroll
  for (int kt = 0; kt < KT_H; ++kt)
    afrag[kt] = load_a(H_all, HID, mt * 16, kt * 32, lane);

  float m[8], s[8];
#pragma unroll
  for (int r = 0; r < 8; ++r) { m[r] = -INFINITY; s[r] = 0.f; }

  for (int j = 0; j < NT_PER_WAVE; ++j) {
    int nt = nt0 + j;
    const _Float16* bp = Wo_pk + (size_t)nt * KT_H * 32 * 16 + lane * 16;
    if (j + 1 < NT_PER_WAVE)   // global_prefetch_b8 of next packed-B block
      __builtin_prefetch(bp + (size_t)KT_H * 32 * 16, 0, 0);
    v8f acc = {};
#pragma unroll
    for (int kt = 0; kt < KT_H; ++kt) {
      v16h b = *(const v16h*)(bp + (size_t)kt * 32 * 16);
      acc = wmma16(afrag[kt], b, acc);
    }
    float bv = bo[nt * 16 + (lane & 15)];
#pragma unroll
    for (int r = 0; r < 8; ++r) {
      float v = acc[r] + bv;
      float nm = fmaxf(m[r], v);
      s[r] = s[r] * __expf(m[r] - nm) + __expf(v - nm);
      m[r] = nm;
    }
  }

  // butterfly combine across the 16 lanes of each half (rows r / r+8)
#pragma unroll
  for (int r = 0; r < 8; ++r) {
    float mm = m[r], sv = s[r];
#pragma unroll
    for (int k = 1; k < 16; k <<= 1)
      sm_merge(mm, sv, __shfl_xor(mm, k, 32), __shfl_xor(sv, k, 32));
    if ((lane & 15) == 0) {
      int rr = ((lane >> 4) << 3) + r;      // row within 16-row tile
      lm[wv][rr] = mm; lsum[wv][rr] = sv;
    }
  }
  __syncthreads();
  if (threadIdx.x < 16) {
    int rr = threadIdx.x;
    float mm = lm[0][rr], sv = lsum[0][rr];
#pragma unroll
    for (int w2 = 1; w2 < 8; ++w2) sm_merge(mm, sv, lm[w2][rr], lsum[w2][rr]);
    int row = mt * 16 + rr;
    pm[(size_t)chunk * ROWS + row] = mm;
    ps[(size_t)chunk * ROWS + row] = sv;
  }
}

// K3b: fold the 25 chunk partials per row -> rmax, 1/rsum (tiny kernel).
__global__ void k_combine(const float* __restrict__ pm, const float* __restrict__ ps,
                          float* __restrict__ rmax, float* __restrict__ rinv) {
  int row = blockIdx.x * blockDim.x + threadIdx.x;
  if (row >= ROWS) return;
  float m = -INFINITY, s = 0.f;
#pragma unroll
  for (int c = 0; c < CHUNKS; ++c)
    sm_merge(m, s, pm[(size_t)c * ROWS + row], ps[(size_t)c * ROWS + row]);
  rmax[row] = m;
  rinv[row] = 1.f / s;
}

// =====================================================================
// K3c: recompute the identical WMMA tile (same kt order -> bitwise same
// acc as the stats pass) and write exp(x - max) * inv directly.
// Single 524 MB output write — the HBM floor for this problem.
// =====================================================================
__global__ void __launch_bounds__(256)
k_logit_write(const _Float16* __restrict__ H_all,
              const _Float16* __restrict__ Wo_pk,
              const float* __restrict__ bo,
              const float* __restrict__ rmax, const float* __restrict__ rinv,
              float* __restrict__ out) {
  int lane = threadIdx.x & 31;
  int mt = blockIdx.y;                              // 0..255
  int nt = blockIdx.x * 8 + (threadIdx.x >> 5);     // 0..1999
  v8f acc = {};
#pragma unroll
  for (int kt = 0; kt < KT_H; ++kt) {
    v16h a = load_a(H_all, HID, mt * 16, kt * 32, lane);
    v16h b = *(const v16h*)(Wo_pk + ((size_t)(nt * KT_H + kt) * 32 + lane) * 16);
    acc = wmma16(a, b, acc);
  }
  int n = nt * 16 + (lane & 15);
  float bv = bo[n];
  int m0 = mt * 16 + ((lane >> 4) << 3);
#pragma unroll
  for (int r = 0; r < 8; ++r) {
    int row = m0 + r;                   // row = t*BATCH + b
    int t = row >> 6, b = row & 63;     // BATCH = 64
    out[(size_t)(b * SEQ + t) * VOCAB + n] =
        __expf(acc[r] + bv - rmax[row]) * rinv[row];
  }
}

// =====================================================================
extern "C" void kernel_launch(void* const* d_in, const int* in_sizes, int n_in,
                              void* d_out, int out_size, void* d_ws, size_t ws_size,
                              hipStream_t stream) {
  (void)in_sizes; (void)n_in; (void)out_size; (void)ws_size;
  const int*   input_x = (const int*)  d_in[0];
  const float* emb     = (const float*)d_in[1];
  const float* Wi  = (const float*)d_in[2],  *Ui  = (const float*)d_in[3],  *bi  = (const float*)d_in[4];
  const float* Wf  = (const float*)d_in[5],  *Uf  = (const float*)d_in[6],  *bf  = (const float*)d_in[7];
  const float* Wog = (const float*)d_in[8],  *Uog = (const float*)d_in[9],  *bog = (const float*)d_in[10];
  const float* Wc  = (const float*)d_in[11], *Uc  = (const float*)d_in[12], *bc  = (const float*)d_in[13];
  const float* Wo  = (const float*)d_in[14], *bo  = (const float*)d_in[15];
  float* out = (float*)d_out;

  // workspace carve-out (~38 MB), 256B aligned slices
  size_t off = 0;
  auto carve = [&](size_t bytes) {
    void* p = (char*)d_ws + off;
    off += (bytes + 255) & ~(size_t)255;
    return p;
  };
  _Float16* XS    = (_Float16*)carve((size_t)ROWS * EMB * 2);
  _Float16* Hall  = (_Float16*)carve((size_t)ROWS * HID * 2);
  float*    gx    = (float*)   carve((size_t)ROWS * GCOLS * 4);
  _Float16* Wx_pk = (_Float16*)carve((size_t)NT_G * KT_X * 32 * 16 * 2);
  _Float16* U_pk  = (_Float16*)carve((size_t)NT_G * KT_H * 32 * 16 * 2);
  _Float16* Wo_pk = (_Float16*)carve((size_t)NT_V * KT_H * 32 * 16 * 2);
  float*    pm    = (float*)   carve((size_t)CHUNKS * ROWS * 4);
  float*    ps    = (float*)   carve((size_t)CHUNKS * ROWS * 4);
  float*    rmax  = (float*)   carve((size_t)ROWS * 4);
  float*    rinv  = (float*)   carve((size_t)ROWS * 4);

  k_embed<<<(ROWS * EMB + 255) / 256, 256, 0, stream>>>(input_x, emb, XS);
  k_pack_gates<<<(NT_G * KT_X * 32 + 255) / 256, 256, 0, stream>>>(Wi, Wf, Wog, Wc, EMB, Wx_pk);
  k_pack_gates<<<(NT_G * KT_H * 32 + 255) / 256, 256, 0, stream>>>(Ui, Uf, Uog, Uc, HID, U_pk);
  k_pack_wo<<<(NT_V * KT_H * 32 + 255) / 256, 256, 0, stream>>>(Wo, Wo_pk);

  // all timesteps' input-side gate pre-activations (time-independent)
  k_gates_x<<<(ROWS / 16) * NT_G / 8, 256, 0, stream>>>(XS, Wx_pk, bi, bf, bog, bc, gx);

  // sequential recurrence on one WGP
  k_recurrence<<<1, 1024, 0, stream>>>(gx, U_pk, Hall);

  // fused projection + softmax: stats (no logit spill) -> combine -> write
  k_logit_stats<<<dim3(CHUNKS, ROWS / 16), 256, 0, stream>>>(Hall, Wo_pk, bo, pm, ps);
  k_combine<<<(ROWS + 255) / 256, 256, 0, stream>>>(pm, ps, rmax, rinv);
  k_logit_write<<<dim3(NT_V / 8, ROWS / 16), 256, 0, stream>>>(Hall, Wo_pk, bo, rmax, rinv, out);
}
